// SNREstimator_25082609009146
// MI455X (gfx1250) — 8-bit matrix operand order repaired and checked by reading, compiled, not run on hardware
//
#include <hip/hip_runtime.h>
#include <hip/hip_bf16.h>
#include <math.h>


#define BB    64
#define FF    257
#define TT    2000
#define FPAD8 320
#define NKT   5
#define NNT   125
#define NMT   5
#define NMEL  80
#define TILE_BYTES 1024
#define A_BYTES (NMT * NKT * TILE_BYTES)
#define WGNT  16

typedef __attribute__((ext_vector_type(8))) int   v8i;
typedef __attribute__((ext_vector_type(8))) float v8f;

__device__ __forceinline__ float sigmoidf_(float x) {
  return 1.0f / (1.0f + __expf(-x));
}

__device__ __forceinline__ float fast_tanhf_(float y) {
  float e = __expf(2.0f * y);
  return 1.0f - 2.0f / (e + 1.0f);
}

__device__ __forceinline__ unsigned int f32_to_fp8_(float x) {
#if __has_builtin(__builtin_amdgcn_cvt_pk_fp8_f32)
  return (unsigned int)__builtin_amdgcn_cvt_pk_fp8_f32(x, x, 0, false) & 0xFFu;
#else
  float ax = fminf(x, 448.0f);
  if (!(ax > 0.0f)) return 0u;
  unsigned int u = __float_as_uint(ax);
  int exp = (int)((u >> 23) & 0xFF) - 127;
  unsigned int mant = u & 0x7FFFFFu;
  if (exp >= -6) {
    unsigned int v = (((unsigned int)(exp + 7)) << 3) +
                     ((mant + (1u << 19)) >> 20);
    return v > 0x7Eu ? 0x7Eu : v;
  }
  if (exp < -10) return 0u;
  int sh = -6 - exp;
  unsigned int full = mant | 0x800000u;
  return (full + (1u << (19 + sh))) >> (20 + sh);
#endif
}


__global__ void snr_aprep_kernel(const float* __restrict__ mel_fb,
                                 unsigned char* __restrict__ aTiles) {
  int idx = blockIdx.x * blockDim.x + threadIdx.x;
  if (idx >= A_BYTES) return;
  int tile = idx >> 10;
  int r    = idx & 1023;
  int mt = tile / NKT;
  int kt = tile % NKT;
  int lane = r >> 5;
  int j    = r & 31;
  int v = j >> 2, s = j & 3;
  int kl = 4 * (v & 1) + 8 * ((v >> 1) & 1) + 32 * (v >> 2) +
           ((lane >= 16) ? 16 : 0) + s;
  int m = mt * 16 + (lane & 15);
  int f = kt * 64 + kl;
  float val = (f < FF) ? mel_fb[m * FF + f] : 0.0f;
  aTiles[idx] = (unsigned char)f32_to_fp8_(val);
}

__global__ void snr_scan_kernel(const float* __restrict__ mag,
                                const float* __restrict__ p_scale,
                                const float* __restrict__ p_beta,
                                const float* __restrict__ p_gamma,
                                unsigned char* __restrict__ snrTiles) {
  int wid = blockIdx.x;
  int b = wid / (FPAD8 / 32);
  int w = wid % (FPAD8 / 32);
  int f = w * 32 + threadIdx.x;
  int kt = f >> 6;
  int k  = f & 63;

  const size_t tileBase0 = (size_t)(b * NKT + kt) * NNT * TILE_BYTES;
  const int g  = (k >> 4) & 1;
  const int hi = (k >> 5) & 1;
  const int j  = 4 * ((k >> 2) & 3) + (k & 3) + 16 * hi;
  const int laneSlot = 16 * g;

  if (f >= FF) {
    for (int t = 0; t < TT; ++t) {
      int nt = t >> 4, n = t & 15;
      snrTiles[tileBase0 + (size_t)nt * TILE_BYTES + (laneSlot + n) * 32 + j] = 0;
    }
    return;
  }

  const float beta  = sigmoidf_(p_beta[0]);
  const float gamma = sigmoidf_(p_gamma[0]);
  const float scale = fabsf(p_scale[0]);

  const float* __restrict__ row = mag + ((size_t)b * FF + f) * TT;

  float init = (row[0] + row[1] + row[2] + row[3] + row[4]) * 0.2f;
  float nf     = init;
  float floorv = init * 0.5f;

  for (int t0 = 0; t0 < TT; t0 += 4) {
    float4 v4 = *(const float4*)(row + t0);
    float vals[4] = {v4.x, v4.y, v4.z, v4.w};
#pragma unroll
    for (int jj = 0; jj < 4; ++jj) {
      float m = vals[jj];
      float a = (m > nf) ? gamma : beta;
      nf = (1.0f - a) * nf + a * m;
      float eff = fmaxf(nf, floorv);
      float snr = m / (scale * eff + 1e-8f);
      int t = t0 + jj;
      int nt = t >> 4, n = t & 15;
      snrTiles[tileBase0 + (size_t)nt * TILE_BYTES + (laneSlot + n) * 32 + j] =
          (unsigned char)f32_to_fp8_(snr);
    }
  }
}

__global__ void __launch_bounds__(256)
snr_wmma_gemm_kernel(const unsigned char* __restrict__ aTiles,
                     const unsigned char* __restrict__ snrTiles,
                     float* __restrict__ out) {
  __shared__ unsigned char aLds[A_BYTES];

  const int tid = threadIdx.x;

  {
    const uint4* __restrict__ src = (const uint4*)aTiles;
    uint4* dst = (uint4*)aLds;
#pragma unroll 4
    for (int i = tid; i < A_BYTES / 16; i += 256) dst[i] = src[i];
  }
  __syncthreads();

  const int wave = tid >> 5;
  const int lane = tid & 31;
  const int b  = blockIdx.x / WGNT;
  const int nt = (blockIdx.x % WGNT) * 8 + wave;
  if (nt >= NNT) return;

  v8i bfrag[NKT];
#pragma unroll
  for (int kt = 0; kt < NKT; ++kt) {
    bfrag[kt] = *(const v8i*)(snrTiles +
        ((size_t)(b * NKT + kt) * NNT + nt) * TILE_BYTES + lane * 32);
  }

  v8f c[NMT];
#pragma unroll
  for (int mt = 0; mt < NMT; ++mt) c[mt] = (v8f){0,0,0,0,0,0,0,0};

#pragma unroll
  for (int kt = 0; kt < NKT; ++kt) {
#pragma unroll
    for (int mt = 0; mt < NMT; ++mt) {
      const v8i afrag =
          *(const v8i*)(aLds + (mt * NKT + kt) * TILE_BYTES + lane * 32);
      c[mt] = __builtin_amdgcn_wmma_f32_16x16x64_fp8_fp8(
          afrag, bfrag[kt], (short)0, c[mt], false, false);
    }
  }

  const int t     = nt * 16 + (lane & 15);
  const int mbase = (lane >= 16) ? 8 : 0;
#pragma unroll
  for (int mt = 0; mt < NMT; ++mt) {
#pragma unroll
    for (int r = 0; r < 8; ++r) {
      int m = mt * 16 + r + mbase;
      out[((size_t)b * NMEL + m) * TT + t] = fast_tanhf_(c[mt][r] * 0.1f);
    }
  }
}

extern "C" void kernel_launch(void* const* d_in, const int* in_sizes, int n_in,
                              void* d_out, int out_size, void* d_ws, size_t ws_size,
                              hipStream_t stream) {
  const float* mag     = (const float*)d_in[0];
  const float* mel_fb  = (const float*)d_in[1];
  const float* p_scale = (const float*)d_in[2];
  const float* p_beta  = (const float*)d_in[3];
  const float* p_gamma = (const float*)d_in[4];
  float* out = (float*)d_out;

  unsigned char* aTiles   = (unsigned char*)d_ws;
  unsigned char* snrTiles = aTiles + A_BYTES;

  snr_aprep_kernel<<<(A_BYTES + 255) / 256, 256, 0, stream>>>(mel_fb, aTiles);
  snr_scan_kernel<<<BB * (FPAD8 / 32), 32, 0, stream>>>(mag, p_scale, p_beta,
                                                        p_gamma, snrTiles);
  snr_wmma_gemm_kernel<<<BB * WGNT, 256, 0, stream>>>(aTiles, snrTiles, out);
}
